// DiTBlock_57612691308676
// MI455X (gfx1250) — compile-verified, matrix operand order checked by reading, not run on hardware
//
#include <hip/hip_runtime.h>
#include <hip/hip_bf16.h>

typedef __attribute__((ext_vector_type(16))) _Float16 v16h;
typedef __attribute__((ext_vector_type(8)))  _Float16 v8h;
typedef __attribute__((ext_vector_type(8)))  float    v8f;

#define EMB 384
#define NH  6
#define HD  64
#define BB  4
#define SS  4096
#define TOK (BB * SS)
#define FF  (4 * EMB)

// ---------------------------------------------------------------------------
// Fragment helpers (wave32, v_wmma_f32_16x16x32_f16)
// A fragment: lane L holds row (row0 + L%16); 16 f16 = two contiguous 16B runs
//   elems 0..7  <-> K = (L/16)*8 + e
//   elems 8..15 <-> K = 16 + (L/16)*8 + (e-8)
// B fragment mirrors A with lane = output column (weights stored [N][K]).
// C/D f32: lane L holds col N = L%16; elem i <-> row M = (L/16)*8 + i.
// ---------------------------------------------------------------------------
__device__ __forceinline__ v16h load_frag(const _Float16* __restrict__ base,
                                          int row0, int ld, int kb, int lane) {
  const int r = lane & 15, hf = lane >> 4;
  const _Float16* p = base + (size_t)(row0 + r) * ld + kb + hf * 8;
  v8h lo = *(const v8h*)p;
  v8h hi = *(const v8h*)(p + 16);
  v16h out;
#pragma unroll
  for (int i = 0; i < 8; ++i) { out[i] = lo[i]; out[8 + i] = hi[i]; }
  return out;
}

__device__ __forceinline__ v8f wmma_f16(v16h a, v16h b, v8f c) {
  return __builtin_amdgcn_wmma_f32_16x16x32_f16(false, a, false, b, (short)0, c,
                                                false, false);
}

__device__ __forceinline__ v8f vzero() {
  v8f z = {0.f, 0.f, 0.f, 0.f, 0.f, 0.f, 0.f, 0.f};
  return z;
}

// 32x64 output tile per wave: two A row-fragments share four B fragments.
// Per k-step: 6 fragment loads feed 8 WMMAs (each B fragment -> 2 WMMAs),
// halving per-WMMA load pressure vs a 16x64 tile.
__device__ __forceinline__ void gemm32x64(const _Float16* __restrict__ A, int lda,
                                          int row0,
                                          const _Float16* __restrict__ Wt, int ldw,
                                          int n0, int K, v8f acc[2][4], int lane) {
  for (int kb = 0; kb < K; kb += 32) {
    v16h a0 = load_frag(A, row0, lda, kb, lane);
    v16h a1 = load_frag(A, row0 + 16, lda, kb, lane);
#pragma unroll
    for (int t = 0; t < 4; ++t) {
      v16h b = load_frag(Wt, n0 + 16 * t, ldw, kb, lane);
      acc[0][t] = wmma_f16(a0, b, acc[0][t]);
      acc[1][t] = wmma_f16(a1, b, acc[1][t]);
    }
  }
}

// ---------------------------------------------------------------------------
// Weight convert+transpose: src f32 [K][N] -> dst f16 [N][K]
// ---------------------------------------------------------------------------
__global__ void k_convert_t(const float* __restrict__ src,
                            _Float16* __restrict__ dst, int K, int N) {
  int idx = blockIdx.x * blockDim.x + threadIdx.x;
  if (idx >= K * N) return;
  int k = idx / N, n = idx % N;
  dst[(size_t)n * K + k] = (_Float16)src[idx];
}

// ---------------------------------------------------------------------------
// AdaLN projections: mods[p][b][n] = cond[b] @ w_p + b_p  (p = g1,be1,a1,g2,be2,a2)
// ---------------------------------------------------------------------------
__global__ void k_adaln(const float* __restrict__ cond,
                        const float* w0, const float* b0, const float* w1, const float* b1,
                        const float* w2, const float* b2, const float* w3, const float* b3,
                        const float* w4, const float* b4, const float* w5, const float* b5,
                        float* __restrict__ mods) {
  int idx = blockIdx.x * blockDim.x + threadIdx.x;
  if (idx >= 6 * BB * EMB) return;
  int p = idx / (BB * EMB), b = (idx / EMB) % BB, n = idx % EMB;
  const float *w, *bi;
  switch (p) {
    case 0: w = w0; bi = b0; break;
    case 1: w = w1; bi = b1; break;
    case 2: w = w2; bi = b2; break;
    case 3: w = w3; bi = b3; break;
    case 4: w = w4; bi = b4; break;
    default: w = w5; bi = b5; break;
  }
  const float* c = cond + b * EMB;
  float acc = bi[n];
  for (int k = 0; k < EMB; ++k) acc += c[k] * w[k * EMB + n];
  mods[idx] = acc;
}

// ---------------------------------------------------------------------------
// LayerNorm + AdaLN modulate -> f16.  One wave per token (12 elems/lane).
// ---------------------------------------------------------------------------
__global__ void k_ln_mod(const float* __restrict__ src, _Float16* __restrict__ dst,
                         const float* __restrict__ lw, const float* __restrict__ lb,
                         const float* __restrict__ gamma,
                         const float* __restrict__ beta) {
  int wave = (blockIdx.x * blockDim.x + threadIdx.x) >> 5;
  int lane = threadIdx.x & 31;
  const float* x = src + (size_t)wave * EMB;
  int b = wave >> 12;  // token / 4096
  float xs[12], s = 0.f, s2 = 0.f;
#pragma unroll
  for (int t = 0; t < 12; ++t) {
    float v = x[lane + 32 * t];
    xs[t] = v; s += v; s2 += v * v;
  }
#pragma unroll
  for (int m = 1; m < 32; m <<= 1) {
    s  += __shfl_xor(s, m, 32);
    s2 += __shfl_xor(s2, m, 32);
  }
  float mean = s * (1.0f / EMB);
  float var  = s2 * (1.0f / EMB) - mean * mean;
  float r = rsqrtf(var + 1e-5f);
  _Float16* d = dst + (size_t)wave * EMB;
#pragma unroll
  for (int t = 0; t < 12; ++t) {
    int j = lane + 32 * t;
    float v = (xs[t] - mean) * r * lw[j] + lb[j];
    v = v * (1.0f + gamma[b * EMB + j]) + beta[b * EMB + j];
    d[j] = (_Float16)v;
  }
}

// ---------------------------------------------------------------------------
// QKV GEMM: y1[TOK,384] @ wqkv_t[1152,384]^T.  q scaled by 1/8; v stored transposed.
// ---------------------------------------------------------------------------
__global__ void k_qkv(const _Float16* __restrict__ y1,
                      const _Float16* __restrict__ wqkv_t,
                      _Float16* __restrict__ q, _Float16* __restrict__ k,
                      _Float16* __restrict__ vt) {
  int wave = (blockIdx.x * blockDim.x + threadIdx.x) >> 5;
  int lane = threadIdx.x & 31;
  int tm = wave % (TOK / 32), tn = wave / (TOK / 32);
  int row0 = tm * 32, n0 = tn * 64;
  v8f acc[2][4] = {{vzero(), vzero(), vzero(), vzero()},
                   {vzero(), vzero(), vzero(), vzero()}};
  gemm32x64(y1, EMB, row0, wqkv_t, EMB, n0, EMB, acc, lane);
  int hf = lane >> 4, col = lane & 15;
#pragma unroll
  for (int rr = 0; rr < 2; ++rr)
#pragma unroll
    for (int t = 0; t < 4; ++t)
#pragma unroll
      for (int i = 0; i < 8; ++i) {
        int m = row0 + rr * 16 + hf * 8 + i;
        int n = n0 + 16 * t + col;
        float val = acc[rr][t][i];
        if (n < EMB) {
          q[(size_t)m * EMB + n] = (_Float16)(val * 0.125f);  // fold 1/sqrt(HD)
        } else if (n < 2 * EMB) {
          k[(size_t)m * EMB + (n - EMB)] = (_Float16)val;
        } else {
          int nn = n - 2 * EMB, h = nn >> 6, d = nn & 63, b = m >> 12, tok = m & 4095;
          vt[((size_t)(b * NH + h) * HD + d) * SS + tok] = (_Float16)val;
        }
      }
}

// ---------------------------------------------------------------------------
// Flash attention, transposed form. One wave per (b, h, 16-query tile).
// S^T = K . Q^T  (queries live in lanes);  O^T += V^T . P^T  (P stays in-lane).
// All 8 fragment loads issue at the top of each iteration so the V^T loads
// overlap with the score WMMAs and the online-softmax VALU work.
// ---------------------------------------------------------------------------
__global__ void k_attn(const _Float16* __restrict__ q,
                       const _Float16* __restrict__ kk,
                       const _Float16* __restrict__ vt,
                       _Float16* __restrict__ out) {
  int wave = (blockIdx.x * blockDim.x + threadIdx.x) >> 5;
  int lane = threadIdx.x & 31;
  int qt = wave & 255;
  int h  = (wave >> 8) % NH;
  int b  = wave / (256 * NH);
  const _Float16* qptr = q  + ((size_t)(b * SS) + qt * 16) * EMB + h * HD;
  const _Float16* kptr = kk + (size_t)(b * SS) * EMB + h * HD;
  const _Float16* vptr = vt + (size_t)((b * NH + h) * HD) * SS;

  v16h bq0 = load_frag(qptr, 0, EMB, 0, lane);   // Q^T fragment, d = 0..31
  v16h bq1 = load_frag(qptr, 0, EMB, 32, lane);  // d = 32..63

  float mi = -__builtin_inff(), li = 0.f;
  v8f o[4] = {vzero(), vzero(), vzero(), vzero()};

  for (int kt = 0; kt < SS; kt += 32) {
    // all loads for this 32-key step up front
    v16h a00 = load_frag(kptr, kt, EMB, 0, lane);
    v16h a10 = load_frag(kptr, kt + 16, EMB, 0, lane);
    v16h a01 = load_frag(kptr, kt, EMB, 32, lane);
    v16h a11 = load_frag(kptr, kt + 16, EMB, 32, lane);
    v16h va0 = load_frag(vptr,  0, SS, kt, lane);  // V^T rows = d, cols = keys
    v16h va1 = load_frag(vptr, 16, SS, kt, lane);
    v16h va2 = load_frag(vptr, 32, SS, kt, lane);
    v16h va3 = load_frag(vptr, 48, SS, kt, lane);

    v8f s0 = vzero(), s1 = vzero();
    s0 = wmma_f16(a00, bq0, s0);
    s1 = wmma_f16(a10, bq0, s1);
    s0 = wmma_f16(a01, bq1, s0);
    s1 = wmma_f16(a11, bq1, s1);

    // per-lane query: online softmax over this lane's 16 j's + partner half
    float mloc = s0[0];
#pragma unroll
    for (int e = 0; e < 8; ++e) {
      mloc = fmaxf(mloc, s0[e]);
      mloc = fmaxf(mloc, s1[e]);
    }
    mloc = fmaxf(mloc, __shfl_xor(mloc, 16, 32));
    float mnew = fmaxf(mi, mloc);
    float corr = __expf(mi - mnew);
    li *= corr;
#pragma unroll
    for (int t = 0; t < 4; ++t)
#pragma unroll
      for (int e = 0; e < 8; ++e) o[t][e] *= corr;

    // P^T in exactly B-fragment order: tile0 -> elems 0..7, tile1 -> 8..15
    v16h p;
#pragma unroll
    for (int e = 0; e < 8; ++e) {
      float p0 = __expf(s0[e] - mnew);
      float p1 = __expf(s1[e] - mnew);
      li += p0 + p1;
      p[e] = (_Float16)p0;
      p[8 + e] = (_Float16)p1;
    }
    mi = mnew;

    o[0] = wmma_f16(va0, p, o[0]);
    o[1] = wmma_f16(va1, p, o[1]);
    o[2] = wmma_f16(va2, p, o[2]);
    o[3] = wmma_f16(va3, p, o[3]);
  }

  li += __shfl_xor(li, 16, 32);
  float inv = 1.0f / li;
  int i = lane & 15, hf = lane >> 4;
  _Float16* ob = out + ((size_t)(b * SS) + qt * 16 + i) * EMB + h * HD;
#pragma unroll
  for (int t = 0; t < 4; ++t)
#pragma unroll
    for (int e = 0; e < 8; ++e)
      ob[t * 16 + hf * 8 + e] = (_Float16)(o[t][e] * inv);
}

// ---------------------------------------------------------------------------
// WO GEMM + alpha1 + residual:  y = x + (att @ wo) * alpha1
// ---------------------------------------------------------------------------
__global__ void k_wo(const _Float16* __restrict__ att,
                     const _Float16* __restrict__ wot,
                     const float* __restrict__ x, const float* __restrict__ alpha1,
                     float* __restrict__ y) {
  int wave = (blockIdx.x * blockDim.x + threadIdx.x) >> 5;
  int lane = threadIdx.x & 31;
  int tm = wave % (TOK / 32), tn = wave / (TOK / 32);
  int row0 = tm * 32, n0 = tn * 64;
  v8f acc[2][4] = {{vzero(), vzero(), vzero(), vzero()},
                   {vzero(), vzero(), vzero(), vzero()}};
  gemm32x64(att, EMB, row0, wot, EMB, n0, EMB, acc, lane);
  int hf = lane >> 4, col = lane & 15;
#pragma unroll
  for (int rr = 0; rr < 2; ++rr)
#pragma unroll
    for (int t = 0; t < 4; ++t)
#pragma unroll
      for (int i = 0; i < 8; ++i) {
        int m = row0 + rr * 16 + hf * 8 + i;
        int n = n0 + 16 * t + col;
        y[(size_t)m * EMB + n] =
            x[(size_t)m * EMB + n] + acc[rr][t][i] * alpha1[(m >> 12) * EMB + n];
      }
}

// ---------------------------------------------------------------------------
// FF1 GEMM + bias + ReLU -> f16
// ---------------------------------------------------------------------------
__global__ void k_ff1(const _Float16* __restrict__ z,
                      const _Float16* __restrict__ w1t,
                      const float* __restrict__ bias, _Float16* __restrict__ hmid) {
  int wave = (blockIdx.x * blockDim.x + threadIdx.x) >> 5;
  int lane = threadIdx.x & 31;
  int tm = wave % (TOK / 32), tn = wave / (TOK / 32);
  int row0 = tm * 32, n0 = tn * 64;
  v8f acc[2][4] = {{vzero(), vzero(), vzero(), vzero()},
                   {vzero(), vzero(), vzero(), vzero()}};
  gemm32x64(z, EMB, row0, w1t, EMB, n0, EMB, acc, lane);
  int hf = lane >> 4, col = lane & 15;
#pragma unroll
  for (int rr = 0; rr < 2; ++rr)
#pragma unroll
    for (int t = 0; t < 4; ++t)
#pragma unroll
      for (int i = 0; i < 8; ++i) {
        int m = row0 + rr * 16 + hf * 8 + i;
        int n = n0 + 16 * t + col;
        float v = acc[rr][t][i] + bias[n];
        hmid[(size_t)m * FF + n] = (_Float16)fmaxf(v, 0.0f);
      }
}

// ---------------------------------------------------------------------------
// FF2 GEMM + bias, * alpha2, + y -> output f32
// ---------------------------------------------------------------------------
__global__ void k_ff2(const _Float16* __restrict__ hmid,
                      const _Float16* __restrict__ w2t,
                      const float* __restrict__ bias, const float* __restrict__ y,
                      const float* __restrict__ alpha2, float* __restrict__ out) {
  int wave = (blockIdx.x * blockDim.x + threadIdx.x) >> 5;
  int lane = threadIdx.x & 31;
  int tm = wave % (TOK / 32), tn = wave / (TOK / 32);
  int row0 = tm * 32, n0 = tn * 64;
  v8f acc[2][4] = {{vzero(), vzero(), vzero(), vzero()},
                   {vzero(), vzero(), vzero(), vzero()}};
  gemm32x64(hmid, FF, row0, w2t, FF, n0, FF, acc, lane);
  int hf = lane >> 4, col = lane & 15;
#pragma unroll
  for (int rr = 0; rr < 2; ++rr)
#pragma unroll
    for (int t = 0; t < 4; ++t)
#pragma unroll
      for (int i = 0; i < 8; ++i) {
        int m = row0 + rr * 16 + hf * 8 + i;
        int n = n0 + 16 * t + col;
        float v = (acc[rr][t][i] + bias[n]) * alpha2[(m >> 12) * EMB + n];
        out[(size_t)m * EMB + n] = y[(size_t)m * EMB + n] + v;
      }
}

// ---------------------------------------------------------------------------
extern "C" void kernel_launch(void* const* d_in, const int* in_sizes, int n_in,
                              void* d_out, int out_size, void* d_ws, size_t ws_size,
                              hipStream_t stream) {
  (void)in_sizes; (void)n_in; (void)out_size; (void)ws_size;
  const float* x     = (const float*)d_in[0];
  const float* cond  = (const float*)d_in[1];
  const float* g1_w  = (const float*)d_in[2];  const float* g1_b  = (const float*)d_in[3];
  const float* be1_w = (const float*)d_in[4];  const float* be1_b = (const float*)d_in[5];
  const float* a1_w  = (const float*)d_in[6];  const float* a1_b  = (const float*)d_in[7];
  const float* g2_w  = (const float*)d_in[8];  const float* g2_b  = (const float*)d_in[9];
  const float* be2_w = (const float*)d_in[10]; const float* be2_b = (const float*)d_in[11];
  const float* a2_w  = (const float*)d_in[12]; const float* a2_b  = (const float*)d_in[13];
  const float* ln1_w = (const float*)d_in[14]; const float* ln1_b = (const float*)d_in[15];
  const float* ln2_w = (const float*)d_in[16]; const float* ln2_b = (const float*)d_in[17];
  const float* wq    = (const float*)d_in[18]; const float* wk    = (const float*)d_in[19];
  const float* wv    = (const float*)d_in[20]; const float* wo    = (const float*)d_in[21];
  const float* ff1_w = (const float*)d_in[22]; const float* ff1_b = (const float*)d_in[23];
  const float* ff2_w = (const float*)d_in[24]; const float* ff2_b = (const float*)d_in[25];
  float* out = (float*)d_out;

  char* ws = (char*)d_ws;
  size_t off = 0;
  auto alloc = [&](size_t bytes) -> char* {
    char* p = ws + off;
    off = (off + bytes + 255) & ~size_t(255);
    return p;
  };
  float*     mods   = (float*)alloc((size_t)6 * BB * EMB * 4);
  _Float16*  wqkv_t = (_Float16*)alloc((size_t)3 * EMB * EMB * 2);
  _Float16*  wo_t   = (_Float16*)alloc((size_t)EMB * EMB * 2);
  _Float16*  ff1_t  = (_Float16*)alloc((size_t)FF * EMB * 2);
  _Float16*  ff2_t  = (_Float16*)alloc((size_t)EMB * FF * 2);
  _Float16*  y1     = (_Float16*)alloc((size_t)TOK * EMB * 2);
  _Float16*  qh     = (_Float16*)alloc((size_t)TOK * EMB * 2);
  _Float16*  kh     = (_Float16*)alloc((size_t)TOK * EMB * 2);
  _Float16*  vt     = (_Float16*)alloc((size_t)TOK * EMB * 2);
  float*     yres   = (float*)alloc((size_t)TOK * EMB * 4);
  _Float16*  hmid   = (_Float16*)alloc((size_t)TOK * FF * 2);
  _Float16*  att    = y1;  // y1 dead after QKV GEMM
  _Float16*  zh     = qh;  // q dead after attention

  // 1) weights -> f16 transposed [N][K]
  k_convert_t<<<(EMB * EMB + 255) / 256, 256, 0, stream>>>(wq, wqkv_t, EMB, EMB);
  k_convert_t<<<(EMB * EMB + 255) / 256, 256, 0, stream>>>(wk, wqkv_t + (size_t)EMB * EMB, EMB, EMB);
  k_convert_t<<<(EMB * EMB + 255) / 256, 256, 0, stream>>>(wv, wqkv_t + (size_t)2 * EMB * EMB, EMB, EMB);
  k_convert_t<<<(EMB * EMB + 255) / 256, 256, 0, stream>>>(wo, wo_t, EMB, EMB);
  k_convert_t<<<(EMB * FF + 255) / 256, 256, 0, stream>>>(ff1_w, ff1_t, EMB, FF);
  k_convert_t<<<(FF * EMB + 255) / 256, 256, 0, stream>>>(ff2_w, ff2_t, FF, EMB);

  // 2) AdaLN conditioning
  k_adaln<<<(6 * BB * EMB + 255) / 256, 256, 0, stream>>>(
      cond, g1_w, g1_b, be1_w, be1_b, a1_w, a1_b, g2_w, g2_b, be2_w, be2_b,
      a2_w, a2_b, mods);
  const float* gamma1 = mods;
  const float* beta1  = mods + 1 * BB * EMB;
  const float* alpha1 = mods + 2 * BB * EMB;
  const float* gamma2 = mods + 3 * BB * EMB;
  const float* beta2  = mods + 4 * BB * EMB;
  const float* alpha2 = mods + 5 * BB * EMB;

  // 3) LN1 + modulate
  k_ln_mod<<<TOK / 8, 256, 0, stream>>>(x, y1, ln1_w, ln1_b, gamma1, beta1);

  // 4) QKV projection (WMMA): (TOK/32)*(1152/64) waves
  k_qkv<<<(TOK / 32) * (3 * EMB / 64) / 8, 256, 0, stream>>>(y1, wqkv_t, qh, kh, vt);

  // 5) flash attention (WMMA)
  k_attn<<<(BB * NH * (SS / 16)) / 8, 256, 0, stream>>>(qh, kh, vt, att);

  // 6) output projection + alpha1 + residual (WMMA)
  k_wo<<<(TOK / 32) * (EMB / 64) / 8, 256, 0, stream>>>(att, wo_t, x, alpha1, yres);

  // 7) LN2 + modulate
  k_ln_mod<<<TOK / 8, 256, 0, stream>>>(yres, zh, ln2_w, ln2_b, gamma2, beta2);

  // 8) FFN (WMMA)
  k_ff1<<<(TOK / 32) * (FF / 64) / 8, 256, 0, stream>>>(zh, ff1_t, ff1_b, hmid);
  k_ff2<<<(TOK / 32) * (EMB / 64) / 8, 256, 0, stream>>>(hmid, ff2_t, ff2_b, yres,
                                                         alpha2, out);
}
